// BEVFormerTiny_67199058313897
// MI455X (gfx1250) — matrix operand order repaired and checked by reading, not compiled, not run on hardware
//
#include <hip/hip_runtime.h>
#include <math.h>

#define Dm   256
#define DFFm 512
#define NHm  8
#define HDm  32
#define NEm  3
#define NDm  6
#define NQm  300
#define NCm  10
#define CSm  10
#define BEVQ 2500
#define NFLT 30450

typedef _Float16 v4h_t  __attribute__((ext_vector_type(4)));
typedef _Float16 v8h_t  __attribute__((ext_vector_type(8)));
typedef _Float16 v16h_t __attribute__((ext_vector_type(16)));
typedef float    v8f_t  __attribute__((ext_vector_type(8)));

__device__ __forceinline__ float wave_sum(float s) {
#pragma unroll
  for (int o = 16; o > 0; o >>= 1) s += __shfl_xor(s, o, 32);
  return s;
}

// ---------------------------------------------------------------------------
// Tiled WMMA GEMM: C[M,N] = act(A[M,K] @ W[K,N] + bias + residual)
// Block = 256 threads = 8 waves (2x4); block tile 64x128, K-step 32.
// Each wave owns a 32x32 output slab: 2 A-fragments x 2 B-fragments,
// 4 wmmas per K-step. A staged row-major (contiguous per-lane runs),
// B staged TRANSPOSED so each lane's 16 k-halves form two contiguous 8-half runs -> b128 loads.
// Edge path: clamped unconditional loads * {0,1} mask (no exec branching).
// ---------------------------------------------------------------------------
__global__ __launch_bounds__(256)
void gemm_wmma(const float* __restrict__ A, int lda,
               const float* __restrict__ W, int ldw,
               const float* __restrict__ bias,
               const float* __restrict__ residual,
               float* __restrict__ C, int ldc,
               int M, int N, int K, int act)
{
  __shared__ _Float16 As[64][40];    // 64 x 32 (pitch 40 halves = 80B, 16B-mult)
  __shared__ _Float16 BsT[128][40];  // transposed: [n][k], 128 x 32

  const int tid  = threadIdx.x;
  const int lane = tid & 31;
  const int wave = tid >> 5;
  const int wm   = wave >> 2;        // 0..1  (M direction, 32 rows each)
  const int wn   = wave & 3;         // 0..3  (N direction, 32 cols each)
  const int bm   = blockIdx.y * 64;
  const int bn   = blockIdx.x * 128;
  const int l16  = lane & 15;
  const int hf   = lane >> 4;        // half-wave index

  v8f_t acc[2][2] = {{{}, {}}, {{}, {}}};

  const bool full = (bm + 64 <= M) && (bn + 128 <= N);

  for (int kt = 0; kt < K; kt += 32) {
    if (full) {
      // ---- fast path: unconditional vector loads ----
      // A tile: 64x32 floats = 512 float4; 2 per thread
#pragma unroll
      for (int i = 0; i < 2; ++i) {
        int idx = tid + i * 256;           // 0..511
        int r = idx >> 3, c4 = idx & 7;
        const float4 v = *(const float4*)(A + (size_t)(bm + r) * lda + kt + c4 * 4);
        v4h_t h;
        h[0] = (_Float16)v.x; h[1] = (_Float16)v.y;
        h[2] = (_Float16)v.z; h[3] = (_Float16)v.w;
        *(v4h_t*)(&As[r][c4 * 4]) = h;
      }
      // B tile: 32x128, transpose in registers; coalesced dword loads
      {
        int n  = tid & 127;                // 0..127
        int kh = (tid >> 7) * 16;          // 0 or 16
#pragma unroll
        for (int q = 0; q < 4; ++q) {
          int k0 = kh + q * 4;
          const float* wp = W + (size_t)(kt + k0) * ldw + bn + n;
          v4h_t h;
          h[0] = (_Float16)wp[0];
          h[1] = (_Float16)wp[(size_t)ldw];
          h[2] = (_Float16)wp[(size_t)2 * ldw];
          h[3] = (_Float16)wp[(size_t)3 * ldw];
          *(v4h_t*)(&BsT[n][k0]) = h;
        }
      }
    } else {
      // ---- edge path: clamped loads, arithmetic {0,1} masking ----
      // (mask-multiply keeps the load unconditionally live -> no exec
      //  predication; clamped addresses always hit valid tensor data)
      float av[8];
#pragma unroll
      for (int i = 0; i < 8; ++i) {
        int idx = tid + i * 256;
        int r = idx >> 5, c = idx & 31;
        int gr = bm + r, gc = kt + c;
        int cr = (gr < M) ? gr : (M - 1);
        int cc = (gc < K) ? gc : (K - 1);
        float mask = (gr < M && gc < K) ? 1.f : 0.f;
        av[i] = A[(size_t)cr * lda + cc] * mask;
      }
#pragma unroll
      for (int i = 0; i < 8; ++i) {
        int idx = tid + i * 256;
        As[idx >> 5][idx & 31] = (_Float16)av[i];
      }
      float bv[16];
#pragma unroll
      for (int i = 0; i < 16; ++i) {
        int idx = tid + i * 256;           // 0..4095
        int r = idx >> 7, c = idx & 127;
        int gr = kt + r, gc = bn + c;
        int cr = (gr < K) ? gr : (K - 1);
        int cc = (gc < N) ? gc : (N - 1);
        float mask = (gr < K && gc < N) ? 1.f : 0.f;
        bv[i] = W[(size_t)cr * ldw + cc] * mask;
      }
#pragma unroll
      for (int i = 0; i < 16; ++i) {
        int idx = tid + i * 256;
        BsT[idx & 127][idx >> 7] = (_Float16)bv[i];
      }
    }
    __syncthreads();

    // ---- fragments: contiguous b128 LDS loads ----
    v16h_t a[2], b[2];
#pragma unroll
    for (int s = 0; s < 2; ++s) {
      const _Float16* rp = &As[wm * 32 + s * 16 + l16][0];
      v8h_t alo = *(const v8h_t*)(rp + hf * 8);
      v8h_t ahi = *(const v8h_t*)(rp + 16 + hf * 8);
      a[s] = __builtin_shufflevector(alo, ahi, 0,1,2,3,4,5,6,7,8,9,10,11,12,13,14,15);
      const _Float16* cp = &BsT[wn * 32 + s * 16 + l16][0];
      v8h_t blo = *(const v8h_t*)(cp + hf * 8);
      v8h_t bhi = *(const v8h_t*)(cp + 16 + hf * 8);
      b[s] = __builtin_shufflevector(blo, bhi, 0,1,2,3,4,5,6,7,8,9,10,11,12,13,14,15);
    }
#pragma unroll
    for (int ms = 0; ms < 2; ++ms)
#pragma unroll
      for (int ns = 0; ns < 2; ++ns)
        acc[ms][ns] = __builtin_amdgcn_wmma_f32_16x16x32_f16(
            false, a[ms], false, b[ns], (short)0, acc[ms][ns], false, false);
    __syncthreads();
  }

  // ---- epilogue: C layout VGPR r -> M = r + 8*half, N = lane%16 ----
  if (full) {
#pragma unroll
    for (int ms = 0; ms < 2; ++ms) {
#pragma unroll
      for (int r = 0; r < 8; ++r) {
        int row = bm + wm * 32 + ms * 16 + hf * 8 + r;
#pragma unroll
        for (int ns = 0; ns < 2; ++ns) {
          int col = bn + wn * 32 + ns * 16 + l16;
          float vv = acc[ms][ns][r];
          if (bias)     vv += bias[col];
          if (residual) vv += residual[(size_t)row * ldc + col];
          if (act)      vv = fmaxf(vv, 0.f);
          C[(size_t)row * ldc + col] = vv;
        }
      }
    }
  } else {
#pragma unroll
    for (int ms = 0; ms < 2; ++ms) {
#pragma unroll
      for (int r = 0; r < 8; ++r) {
        int row = bm + wm * 32 + ms * 16 + hf * 8 + r;
        if (row >= M) continue;
#pragma unroll
        for (int ns = 0; ns < 2; ++ns) {
          int col = bn + wn * 32 + ns * 16 + l16;
          if (col >= N) continue;
          float vv = acc[ms][ns][r];
          if (bias)     vv += bias[col];
          if (residual) vv += residual[(size_t)row * ldc + col];
          if (act)      vv = fmaxf(vv, 0.f);
          C[(size_t)row * ldc + col] = vv;
        }
      }
    }
  }
}

// ---------------------------------------------------------------------------
// Online-softmax attention: one wave per (query, head); lane = channel (HD=32).
// ---------------------------------------------------------------------------
__global__ __launch_bounds__(256)
void mha_flash(const float* __restrict__ Q, int qs,
               const float* __restrict__ K, int ks,
               const float* __restrict__ V, int vs,
               float* __restrict__ O, int os,
               int Lq, int Lk, float scale)
{
  int wid  = (blockIdx.x * blockDim.x + threadIdx.x) >> 5;
  int lane = threadIdx.x & 31;
  int q = wid / NHm, h = wid % NHm;
  if (q >= Lq) return;
  float qd = Q[(size_t)q * qs + h * HDm + lane];
  float m = -1e30f, l = 0.f, acc = 0.f;
  for (int k = 0; k < Lk; ++k) {
    float s = wave_sum(qd * K[(size_t)k * ks + h * HDm + lane]) * scale;
    float mn = fmaxf(m, s);
    float co = __expf(m - mn), p = __expf(s - mn);
    l   = l * co + p;
    acc = acc * co + p * V[(size_t)k * vs + h * HDm + lane];
    m = mn;
  }
  O[(size_t)q * os + h * HDm + lane] = acc / l;
}

// ---------------------------------------------------------------------------
// Row LayerNorm over D=256: one wave per row, 8 elements per lane.
// ---------------------------------------------------------------------------
__global__ __launch_bounds__(256)
void ln_rows(const float* __restrict__ X, const float* __restrict__ S,
             const float* __restrict__ Bb, float* __restrict__ Y, int M, int act)
{
  int row  = (blockIdx.x * blockDim.x + threadIdx.x) >> 5;
  int lane = threadIdx.x & 31;
  if (row >= M) return;
  const float* x = X + (size_t)row * Dm;
  float v[8], s = 0.f;
#pragma unroll
  for (int i = 0; i < 8; ++i) { v[i] = x[lane + i * 32]; s += v[i]; }
  float mean = wave_sum(s) * (1.f / Dm);
  float var = 0.f;
#pragma unroll
  for (int i = 0; i < 8; ++i) { float d = v[i] - mean; var += d * d; }
  var = wave_sum(var) * (1.f / Dm);
  float inv = rsqrtf(var + 1e-5f);
#pragma unroll
  for (int i = 0; i < 8; ++i) {
    int c = lane + i * 32;
    float o = (v[i] - mean) * inv * S[c] + Bb[c];
    if (act) o = fmaxf(o, 0.f);
    Y[(size_t)row * Dm + c] = o;
  }
}

// ---------------------------------------------------------------------------
// (D,HW) -> (HW,D) transpose + level embedding, LDS 32x32 tile.
// ---------------------------------------------------------------------------
__global__ __launch_bounds__(256)
void flatten_lvl(const float* __restrict__ F, const float* __restrict__ lvl,
                 float* __restrict__ Out, int HW, int startRow)
{
  __shared__ float tile[32][33];
  int p0 = blockIdx.x * 32;
  int d0 = blockIdx.y * 32;
  int tx = threadIdx.x & 31, ty = threadIdx.x >> 5;
#pragma unroll
  for (int r = 0; r < 4; ++r) {
    int pp = p0 + tx;
    tile[ty + 8 * r][tx] = (pp < HW) ? F[(size_t)(d0 + ty + 8 * r) * HW + pp] : 0.f;
  }
  __syncthreads();
#pragma unroll
  for (int r = 0; r < 4; ++r) {
    int pp = p0 + ty + 8 * r;
    int dd = d0 + tx;
    if (pp < HW)
      Out[(size_t)(startRow + pp) * Dm + dd] = tile[tx][ty + 8 * r] + lvl[dd];
  }
}

__global__ void add2_k(const float* a, const float* b, float* o, int n) {
  int i = blockIdx.x * blockDim.x + threadIdx.x;
  if (i < n) o[i] = a[i] + b[i];
}

__global__ void tinit_k(const float* __restrict__ qry, float* __restrict__ t) {
  int i = blockIdx.x * blockDim.x + threadIdx.x;
  if (i < NQm * Dm) {
    int q = i / Dm, d = i % Dm;
    t[i] = qry[(size_t)q * 2 * Dm + d] + qry[(size_t)q * 2 * Dm + Dm + d];
  }
}

__global__ void softmax12_k(float* a, int Lq) {
  int t = blockIdx.x * blockDim.x + threadIdx.x;
  if (t >= Lq * NHm) return;
  float* p = a + (size_t)(t / NHm) * 96 + (t % NHm) * 12;
  float m = -1e30f;
#pragma unroll
  for (int i = 0; i < 12; ++i) m = fmaxf(m, p[i]);
  float e[12], s = 0.f;
#pragma unroll
  for (int i = 0; i < 12; ++i) { e[i] = __expf(p[i] - m); s += e[i]; }
  float inv = 1.f / s;
#pragma unroll
  for (int i = 0; i < 12; ++i) p[i] = e[i] * inv;
}

// ---------------------------------------------------------------------------
// Multi-scale deformable sampling: one wave per (query, head); lane = channel.
// ---------------------------------------------------------------------------
__global__ __launch_bounds__(256)
void msda_sample(const float* __restrict__ V, const float* __restrict__ OFF,
                 const float* __restrict__ ATT, float* __restrict__ Out)
{
  int wid  = (blockIdx.x * blockDim.x + threadIdx.x) >> 5;
  int lane = threadIdx.x & 31;
  if (wid >= BEVQ * NHm) return;
  int q = wid >> 3, h = wid & 7;
  const int Hs[3] = {116, 58, 29};
  const int Ws[3] = {200, 100, 50};
  const int St[3] = {0, 23200, 29000};
  float rx = ((q % 50) + 0.5f) / 50.f;
  float ry = ((q / 50) + 0.5f) / 50.f;
  const float* offq = OFF + (size_t)q * 192 + h * 24;
  const float* attq = ATT + (size_t)q * 96 + h * 12;
  float acc = 0.f;
  for (int l = 0; l < 3; ++l) {
    int Hh = Hs[l], Ww = Ws[l], st = St[l];
    for (int p = 0; p < 4; ++p) {
      float ox = offq[(l * 4 + p) * 2 + 0];
      float oy = offq[(l * 4 + p) * 2 + 1];
      float aw = attq[l * 4 + p];
      float x = (rx + ox / (float)Ww) * (float)Ww - 0.5f;
      float y = (ry + oy / (float)Hh) * (float)Hh - 0.5f;
      float xf = floorf(x), yf = floorf(y);
      int x0 = (int)xf, y0 = (int)yf;
      float fx = x - xf, fy = y - yf;
      float w00 = (1.f - fx) * (1.f - fy), w10 = fx * (1.f - fy);
      float w01 = (1.f - fx) * fy,         w11 = fx * fy;
      float s = 0.f;
      if (x0 >= 0 && x0 < Ww && y0 >= 0 && y0 < Hh)
        s += w00 * V[(size_t)(st + y0 * Ww + x0) * Dm + h * HDm + lane];
      if (x0 + 1 >= 0 && x0 + 1 < Ww && y0 >= 0 && y0 < Hh)
        s += w10 * V[(size_t)(st + y0 * Ww + x0 + 1) * Dm + h * HDm + lane];
      if (x0 >= 0 && x0 < Ww && y0 + 1 >= 0 && y0 + 1 < Hh)
        s += w01 * V[(size_t)(st + (y0 + 1) * Ww + x0) * Dm + h * HDm + lane];
      if (x0 + 1 >= 0 && x0 + 1 < Ww && y0 + 1 >= 0 && y0 + 1 < Hh)
        s += w11 * V[(size_t)(st + (y0 + 1) * Ww + x0 + 1) * Dm + h * HDm + lane];
      acc += aw * s;
    }
  }
  Out[(size_t)q * Dm + h * HDm + lane] = acc;
}

__global__ void sigmoid3_k(float* r) {
  int i = blockIdx.x * blockDim.x + threadIdx.x;
  if (i < NQm * CSm) {
    int c = i % CSm;
    if (c < 3) r[i] = 1.f / (1.f + __expf(-r[i]));
  }
}

// ---------------------------------------------------------------------------
extern "C" void kernel_launch(void* const* d_in, const int* in_sizes, int n_in,
                              void* d_out, int out_size, void* d_ws, size_t ws_size,
                              hipStream_t stream)
{
  (void)in_sizes; (void)n_in; (void)out_size; (void)ws_size;
  const float* feat0    = (const float*)d_in[0];
  const float* feat1    = (const float*)d_in[1];
  const float* feat2    = (const float*)d_in[2];
  const float* lvl_emb  = (const float*)d_in[3];
  const float* bev_emb  = (const float*)d_in[4];
  const float* bev_pos  = (const float*)d_in[5];
  const float* e_sa_inw = (const float*)d_in[6];
  const float* e_sa_inb = (const float*)d_in[7];
  const float* e_sa_ow  = (const float*)d_in[8];
  const float* e_sa_ob  = (const float*)d_in[9];
  const float* e_n1s    = (const float*)d_in[10];
  const float* e_n1b    = (const float*)d_in[11];
  const float* e_ca_vw  = (const float*)d_in[12];
  const float* e_ca_vb  = (const float*)d_in[13];
  const float* e_ca_offw= (const float*)d_in[14];
  const float* e_ca_offb= (const float*)d_in[15];
  const float* e_ca_aw  = (const float*)d_in[16];
  const float* e_ca_ab  = (const float*)d_in[17];
  const float* e_ca_pw  = (const float*)d_in[18];
  const float* e_ca_pb  = (const float*)d_in[19];
  const float* e_n2s    = (const float*)d_in[20];
  const float* e_n2b    = (const float*)d_in[21];
  const float* e_f1w    = (const float*)d_in[22];
  const float* e_f1b    = (const float*)d_in[23];
  const float* e_f2w    = (const float*)d_in[24];
  const float* e_f2b    = (const float*)d_in[25];
  const float* e_n3s    = (const float*)d_in[26];
  const float* e_n3b    = (const float*)d_in[27];
  const float* d_sa_inw = (const float*)d_in[28];
  const float* d_sa_inb = (const float*)d_in[29];
  const float* d_sa_ow  = (const float*)d_in[30];
  const float* d_sa_ob  = (const float*)d_in[31];
  const float* d_ca_inw = (const float*)d_in[32];
  const float* d_ca_inb = (const float*)d_in[33];
  const float* d_ca_ow  = (const float*)d_in[34];
  const float* d_ca_ob  = (const float*)d_in[35];
  const float* d_n1s    = (const float*)d_in[36];
  const float* d_n1b    = (const float*)d_in[37];
  const float* d_n2s    = (const float*)d_in[38];
  const float* d_n2b    = (const float*)d_in[39];
  const float* d_n3s    = (const float*)d_in[40];
  const float* d_n3b    = (const float*)d_in[41];
  const float* d_f1w    = (const float*)d_in[42];
  const float* d_f1b    = (const float*)d_in[43];
  const float* d_f2w    = (const float*)d_in[44];
  const float* d_f2b    = (const float*)d_in[45];
  const float* qry_emb  = (const float*)d_in[46];
  const float* cls_w1   = (const float*)d_in[47];
  const float* cls_b1   = (const float*)d_in[48];
  const float* cls_lns  = (const float*)d_in[49];
  const float* cls_lnb  = (const float*)d_in[50];
  const float* cls_w2   = (const float*)d_in[51];
  const float* cls_b2   = (const float*)d_in[52];
  const float* reg_w1   = (const float*)d_in[53];
  const float* reg_b1   = (const float*)d_in[54];
  const float* reg_w2   = (const float*)d_in[55];
  const float* reg_b2   = (const float*)d_in[56];
  const float* reg_w3   = (const float*)d_in[57];
  const float* reg_b3   = (const float*)d_in[58];
  float* out = (float*)d_out;

  // bump allocator on workspace
  char* base = (char*)d_ws;
  size_t woff = 0;
  auto alloc = [&](size_t nfloats) -> float* {
    float* p = (float*)(base + woff);
    woff += ((nfloats * sizeof(float)) + 255) & ~(size_t)255;
    return p;
  };
  float* flat  = alloc((size_t)NFLT * Dm);
  float* vbuf  = alloc((size_t)NFLT * Dm);
  float* x     = alloc((size_t)BEVQ * Dm);
  float* q2    = alloc((size_t)BEVQ * Dm);
  float* qk    = alloc((size_t)BEVQ * 2 * Dm);
  float* vsm   = alloc((size_t)BEVQ * Dm);
  float* attno = alloc((size_t)BEVQ * Dm);
  float* offb  = alloc((size_t)BEVQ * 192);
  float* attb  = alloc((size_t)BEVQ * 96);
  float* msdao = alloc((size_t)BEVQ * Dm);
  float* ffh   = alloc((size_t)BEVQ * DFFm);
  float* tmp   = alloc((size_t)BEVQ * Dm);
  float* t     = alloc((size_t)NQm * Dm);
  float* dqkv  = alloc((size_t)NQm * 3 * Dm);
  float* dkv   = alloc((size_t)BEVQ * 2 * Dm);
  float* dattn = alloc((size_t)NQm * Dm);
  float* dffh  = alloc((size_t)NQm * DFFm);
  float* u     = alloc((size_t)NQm * Dm);
  float* r1    = alloc((size_t)NQm * Dm);

  auto gemm = [&](const float* A, int lda, const float* W, int ldw,
                  const float* bias, const float* res, float* C, int ldc,
                  int M, int N, int K, int act) {
    dim3 g((N + 127) / 128, (M + 63) / 64);
    gemm_wmma<<<g, 256, 0, stream>>>(A, lda, W, ldw, bias, res, C, ldc, M, N, K, act);
  };
  auto mha = [&](const float* Q, int qs, const float* K, int ks, const float* V, int vs,
                 float* O, int os, int Lq, int Lk) {
    int waves = Lq * NHm;
    mha_flash<<<(waves * 32 + 255) / 256, 256, 0, stream>>>(Q, qs, K, ks, V, vs, O, os, Lq, Lk,
                                                            0.176776695f /*1/sqrt(32)*/);
  };
  auto ln = [&](const float* X, const float* S, const float* B, float* Y, int M, int act) {
    ln_rows<<<(M + 7) / 8, 256, 0, stream>>>(X, S, B, Y, M, act);
  };
  auto add2 = [&](const float* a, const float* b, float* o, int n) {
    add2_k<<<(n + 255) / 256, 256, 0, stream>>>(a, b, o, n);
  };

  // ---- flatten multi-scale features (+ level embeddings) ----
  flatten_lvl<<<dim3((23200 + 31) / 32, 8), 256, 0, stream>>>(feat0, lvl_emb + 0 * Dm, flat, 23200, 0);
  flatten_lvl<<<dim3((5800  + 31) / 32, 8), 256, 0, stream>>>(feat1, lvl_emb + 1 * Dm, flat, 5800, 23200);
  flatten_lvl<<<dim3((1450  + 31) / 32, 8), 256, 0, stream>>>(feat2, lvl_emb + 2 * Dm, flat, 1450, 29000);

  hipMemcpyAsync(x, bev_emb, (size_t)BEVQ * Dm * sizeof(float), hipMemcpyDeviceToDevice, stream);

  // ================= BEV encoder =================
  for (int i = 0; i < NEm; ++i) {
    const float* inw = e_sa_inw + (size_t)i * Dm * 3 * Dm;
    const float* inb = e_sa_inb + (size_t)i * 3 * Dm;
    add2(x, bev_pos, q2, BEVQ * Dm);
    gemm(q2, Dm, inw,            3 * Dm, inb,            nullptr, qk,  2 * Dm, BEVQ, 2 * Dm, Dm, 0);
    gemm(x,  Dm, inw + 2 * Dm,   3 * Dm, inb + 2 * Dm,   nullptr, vsm, Dm,     BEVQ, Dm,     Dm, 0);
    mha(qk, 2 * Dm, qk + Dm, 2 * Dm, vsm, Dm, attno, Dm, BEVQ, BEVQ);
    gemm(attno, Dm, e_sa_ow + (size_t)i * Dm * Dm, Dm, e_sa_ob + (size_t)i * Dm, x, tmp, Dm, BEVQ, Dm, Dm, 0);
    ln(tmp, e_n1s + (size_t)i * Dm, e_n1b + (size_t)i * Dm, x, BEVQ, 0);

    // ---- multi-scale deformable attention ----
    add2(x, bev_pos, q2, BEVQ * Dm);
    gemm(flat, Dm, e_ca_vw + (size_t)i * Dm * Dm,    Dm,  e_ca_vb   + (size_t)i * Dm,  nullptr, vbuf, Dm,  NFLT, Dm,  Dm, 0);
    gemm(q2,   Dm, e_ca_offw + (size_t)i * Dm * 192, 192, e_ca_offb + (size_t)i * 192, nullptr, offb, 192, BEVQ, 192, Dm, 0);
    gemm(q2,   Dm, e_ca_aw  + (size_t)i * Dm * 96,   96,  e_ca_ab   + (size_t)i * 96,  nullptr, attb, 96,  BEVQ, 96,  Dm, 0);
    softmax12_k<<<(BEVQ * NHm + 255) / 256, 256, 0, stream>>>(attb, BEVQ);
    msda_sample<<<(BEVQ * NHm * 32 + 255) / 256, 256, 0, stream>>>(vbuf, offb, attb, msdao);
    gemm(msdao, Dm, e_ca_pw + (size_t)i * Dm * Dm, Dm, e_ca_pb + (size_t)i * Dm, x, tmp, Dm, BEVQ, Dm, Dm, 0);
    ln(tmp, e_n2s + (size_t)i * Dm, e_n2b + (size_t)i * Dm, x, BEVQ, 0);

    // ---- FFN ----
    gemm(x,   Dm,   e_f1w + (size_t)i * Dm * DFFm, DFFm, e_f1b + (size_t)i * DFFm, nullptr, ffh, DFFm, BEVQ, DFFm, Dm,   1);
    gemm(ffh, DFFm, e_f2w + (size_t)i * DFFm * Dm, Dm,   e_f2b + (size_t)i * Dm,   x,       tmp, Dm,   BEVQ, Dm,   DFFm, 0);
    ln(tmp, e_n3s + (size_t)i * Dm, e_n3b + (size_t)i * Dm, x, BEVQ, 0);
  }
  // memory == x

  // ================= DETR decoder =================
  tinit_k<<<(NQm * Dm + 255) / 256, 256, 0, stream>>>(qry_emb, t);
  for (int i = 0; i < NDm; ++i) {
    const float* sinw = d_sa_inw + (size_t)i * Dm * 3 * Dm;
    const float* sinb = d_sa_inb + (size_t)i * 3 * Dm;
    gemm(t, Dm, sinw, 3 * Dm, sinb, nullptr, dqkv, 3 * Dm, NQm, 3 * Dm, Dm, 0);
    mha(dqkv, 3 * Dm, dqkv + Dm, 3 * Dm, dqkv + 2 * Dm, 3 * Dm, dattn, Dm, NQm, NQm);
    gemm(dattn, Dm, d_sa_ow + (size_t)i * Dm * Dm, Dm, d_sa_ob + (size_t)i * Dm, t, tmp, Dm, NQm, Dm, Dm, 0);
    ln(tmp, d_n1s + (size_t)i * Dm, d_n1b + (size_t)i * Dm, t, NQm, 0);

    const float* cinw = d_ca_inw + (size_t)i * Dm * 3 * Dm;
    const float* cinb = d_ca_inb + (size_t)i * 3 * Dm;
    gemm(t, Dm, cinw,      3 * Dm, cinb,      nullptr, dqkv, Dm,     NQm,  Dm,     Dm, 0);
    gemm(x, Dm, cinw + Dm, 3 * Dm, cinb + Dm, nullptr, dkv,  2 * Dm, BEVQ, 2 * Dm, Dm, 0);
    mha(dqkv, Dm, dkv, 2 * Dm, dkv + Dm, 2 * Dm, dattn, Dm, NQm, BEVQ);
    gemm(dattn, Dm, d_ca_ow + (size_t)i * Dm * Dm, Dm, d_ca_ob + (size_t)i * Dm, t, tmp, Dm, NQm, Dm, Dm, 0);
    ln(tmp, d_n2s + (size_t)i * Dm, d_n2b + (size_t)i * Dm, t, NQm, 0);

    gemm(t,    Dm,   d_f1w + (size_t)i * Dm * DFFm, DFFm, d_f1b + (size_t)i * DFFm, nullptr, dffh, DFFm, NQm, DFFm, Dm,   1);
    gemm(dffh, DFFm, d_f2w + (size_t)i * DFFm * Dm, Dm,   d_f2b + (size_t)i * Dm,   t,       tmp,  Dm,   NQm, Dm,   DFFm, 0);
    ln(tmp, d_n3s + (size_t)i * Dm, d_n3b + (size_t)i * Dm, t, NQm, 0);
  }

  // ================= heads =================
  gemm(t, Dm, cls_w1, Dm, cls_b1, nullptr, u, Dm, NQm, Dm, Dm, 0);
  ln(u, cls_lns, cls_lnb, r1, NQm, 1);
  gemm(r1, Dm, cls_w2, NCm, cls_b2, nullptr, out, NCm, NQm, NCm, Dm, 0);

  gemm(t,  Dm, reg_w1, Dm,  reg_b1, nullptr, u,  Dm, NQm, Dm, Dm, 1);
  gemm(u,  Dm, reg_w2, Dm,  reg_b2, nullptr, r1, Dm, NQm, Dm, Dm, 1);
  gemm(r1, Dm, reg_w3, CSm, reg_b3, nullptr, out + NQm * NCm, CSm, NQm, CSm, Dm, 0);
  sigmoid3_k<<<(NQm * CSm + 255) / 256, 256, 0, stream>>>(out + NQm * NCm);
}
